// BinarizedConvNet_40407052320988
// MI455X (gfx1250) — matrix operand order repaired and checked by reading, not compiled, not run on hardware
//
#include <hip/hip_runtime.h>
#include <hip/hip_bf16.h>

typedef __attribute__((ext_vector_type(16))) _Float16 v16h;
typedef __attribute__((ext_vector_type(8)))  _Float16 v8h;
typedef __attribute__((ext_vector_type(8)))  float    v8f;

// ---------------------------------------------------------------------------
// Weight pre-pack: binarize (sign) and lay out in WMMA B-fragment order,
// with a k-remap so the GEMM's k-order can be channel-fastest (NHWC im2col).
//   wp[(((nt*KT)+kt)*32 + lane)*16 + i] = sign(w[n][korig])
//   n = nt*16 + (lane&15),  k = kt*32 + (i>=8)*16 + (lane>=16)*8 + (i&7)
//   korig = (k % C2)*(K/C2) + (k / C2)      (C2==1 -> identity)
// GEMM then loads one B fragment as 32 contiguous bytes per lane.
// ---------------------------------------------------------------------------
__global__ __launch_bounds__(256) void pack_w_kernel(
    const float* __restrict__ w, _Float16* __restrict__ wp, int N, int K, int C2)
{
    const int t = blockIdx.x * blockDim.x + threadIdx.x;
    if (t >= N * K) return;
    const int KT    = K >> 5;
    const int i     = t & 15;
    const int lane  = (t >> 4) & 31;
    const int kt    = (t >> 9) % KT;
    const int nt    = (t >> 9) / KT;
    const int n     = (nt << 4) + (lane & 15);
    const int k     = (kt << 5) + ((i >> 3) << 4) + ((lane >> 4) << 3) + (i & 7);
    const int korig = (k % C2) * (K / C2) + (k / C2);
    const float v   = w[(size_t)n * K + korig];
    wp[t] = (_Float16)((v > 0.0f) ? 1.0f : ((v < 0.0f) ? -1.0f : 0.0f));
}

// ---------------------------------------------------------------------------
// A-fragment gather (CDNA5 16-bit A 16x32 layout): lane holds row m = lane&15,
// element i -> k = k0 + (i>=8)*16 + hgrp*8 + (i&7).  Each 8-element run is
// k-contiguous.  f16 layers use NHWC activations with k = (kh*KW+kw)*C + c,
// so every run is 8 contiguous 16B-aligned halves -> one global_load_b128.
// ---------------------------------------------------------------------------
template<bool XF32, int C, int HH, int WW, int KH, int KW>
__device__ __forceinline__ v16h gather_a(const void* rowBase, int ih0, int iw0,
                                         int k0, int hgrp, bool ok)
{
    constexpr int KHKW = KH * KW;
    constexpr int HW   = HH * WW;
    v16h a;
    if (!ok) {
#pragma unroll
        for (int i = 0; i < 16; ++i) a[i] = (_Float16)0.0f;
        return a;
    }
#pragma unroll
    for (int r = 0; r < 2; ++r) {
        const int ks = k0 + r * 16 + (hgrp << 3);   // 8-aligned k-run start
        if constexpr (XF32 && KW == 8 && (KHKW % 8 == 0)) {
            // conv1 (NCHW f32 input, k = c*KHKW + kh*8 + kw):
            // run = one full kw row, 16B-aligned contiguous f32.
            const float* xb = (const float*)rowBase;
            const int c  = ks / KHKW;
            const int kh = (ks - c * KHKW) >> 3;
            const float* p = xb + (size_t)c * HW + (ih0 + kh) * WW + iw0;
            const float4 lo = *(const float4*)(p);
            const float4 hi = *(const float4*)(p + 4);
            a[r * 8 + 0] = (_Float16)lo.x; a[r * 8 + 1] = (_Float16)lo.y;
            a[r * 8 + 2] = (_Float16)lo.z; a[r * 8 + 3] = (_Float16)lo.w;
            a[r * 8 + 4] = (_Float16)hi.x; a[r * 8 + 5] = (_Float16)hi.y;
            a[r * 8 + 6] = (_Float16)hi.z; a[r * 8 + 7] = (_Float16)hi.w;
        } else if constexpr (!XF32 && (C % 8 == 0)) {
            // NHWC f16 activations, k = khkw*C + c: run = 8 contiguous halves.
            const _Float16* xb = (const _Float16*)rowBase;
            const int khkw = ks / C;
            const int c0   = ks - khkw * C;
            const int kh   = khkw / KW;
            const int kw   = khkw - kh * KW;
            const _Float16* p = xb + ((size_t)(ih0 + kh) * WW + (iw0 + kw)) * C + c0;
            const v8h v = *(const v8h*)p;
#pragma unroll
            for (int j = 0; j < 8; ++j) a[r * 8 + j] = v[j];
        } else if constexpr (XF32) {
            // generic f32 scalar fallback (unused by current layers)
            const float* xb = (const float*)rowBase;
#pragma unroll
            for (int j = 0; j < 8; ++j) {
                const int k  = ks + j;
                const int c  = k / KHKW;
                const int rm = k - c * KHKW;
                const int kh = rm / KW;
                const int kw = rm - kh * KW;
                a[r * 8 + j] = (_Float16)xb[(size_t)c * HW + (ih0 + kh) * WW + (iw0 + kw)];
            }
        } else {
            // generic NHWC f16 scalar fallback (unused by current layers)
            const _Float16* xb = (const _Float16*)rowBase;
#pragma unroll
            for (int j = 0; j < 8; ++j) {
                const int k    = ks + j;
                const int khkw = k / C;
                const int c    = k - khkw * C;
                const int kh   = khkw / KW;
                const int kw   = khkw - kh * KW;
                a[r * 8 + j] = xb[((size_t)(ih0 + kh) * WW + (iw0 + kw)) * C + c];
            }
        }
    }
    return a;
}

// ---------------------------------------------------------------------------
// Binarized implicit-GEMM conv / FC.  M = B*OH*OW, N = OC, K = C*KH*KW.
// One wave computes a 32(M) x 32(N) tile: 2 A fragments x 2 B fragments
// -> 4x V_WMMA_F32_16X16X32_F16 per 32-deep K step (2x reuse of each frag).
// All hot-loop traffic is b128 loads (pre-packed B, NHWC/vectorized A).
// ---------------------------------------------------------------------------
template<bool XF32, int C, int HH, int WW, int KH, int KW, int S, int OH, int OW, int OC>
__global__ __launch_bounds__(256) void binconv_wmma(
    const void* __restrict__ xv, const _Float16* __restrict__ wpack,
    const float* __restrict__ bias, float* __restrict__ y, int B)
{
    constexpr int K    = C * KH * KW;
    constexpr int KT   = K / 32;
    constexpr int OHOW = OH * OW;
    constexpr int HW   = HH * WW;
    static_assert(K % 32 == 0, "K must be a multiple of 32");

    const int lane = threadIdx.x & 31;
    const int wave = threadIdx.x >> 5;
    const int hgrp = lane >> 4;
    const int ml   = lane & 15;
    const int mBase = (blockIdx.x * (blockDim.x >> 5) + wave) * 32;
    const int nBase = blockIdx.y * 32;
    const int M = B * OHOW;

    // Row coordinates for the two A fragments this lane feeds.
    int   ih0[2], iw0[2];
    const void* rb[2];
    bool  ok[2];
#pragma unroll
    for (int f = 0; f < 2; ++f) {
        const int m = mBase + f * 16 + ml;
        ok[f] = (m < M);
        const int bb = ok[f] ? (m / OHOW) : 0;
        const int rr = ok[f] ? (m - bb * OHOW) : 0;
        const int oh = rr / OW;
        const int ow = rr - oh * OW;
        ih0[f] = oh * S;
        iw0[f] = ow * S;
        if constexpr (XF32) rb[f] = (const void*)((const float*)xv + (size_t)bb * C * HW);
        else                rb[f] = (const void*)((const _Float16*)xv + (size_t)bb * C * HW);
    }

    const _Float16* wp0 = wpack + ((size_t)(nBase >> 4) * KT) * 512 + lane * 16;
    const _Float16* wp1 = wp0 + (size_t)KT * 512;

    v8f acc00 = {}, acc01 = {}, acc10 = {}, acc11 = {};

    for (int kt = 0; kt < KT; ++kt) {
        const v16h a0 = gather_a<XF32, C, HH, WW, KH, KW>(rb[0], ih0[0], iw0[0], kt * 32, hgrp, ok[0]);
        const v16h a1 = gather_a<XF32, C, HH, WW, KH, KW>(rb[1], ih0[1], iw0[1], kt * 32, hgrp, ok[1]);
        const v16h b0 = *(const v16h*)(wp0 + (size_t)kt * 512);
        const v16h b1 = *(const v16h*)(wp1 + (size_t)kt * 512);
        if (kt + 1 < KT) {
            __builtin_prefetch((const void*)(wp0 + (size_t)(kt + 1) * 512), 0, 1);
            __builtin_prefetch((const void*)(wp1 + (size_t)(kt + 1) * 512), 0, 1);
        }
        acc00 = __builtin_amdgcn_wmma_f32_16x16x32_f16(false, a0, false, b0, (short)0, acc00, false, false);
        acc01 = __builtin_amdgcn_wmma_f32_16x16x32_f16(false, a0, false, b1, (short)0, acc01, false, false);
        acc10 = __builtin_amdgcn_wmma_f32_16x16x32_f16(false, a1, false, b0, (short)0, acc10, false, false);
        acc11 = __builtin_amdgcn_wmma_f32_16x16x32_f16(false, a1, false, b1, (short)0, acc11, false, false);
    }

    // Store D (+bias) to NCHW pre-BN f32 buffer (stats kernel consumes NCHW).
    const int   n0   = nBase + ml;
    const float bia0 = bias[n0];
    const float bia1 = bias[n0 + 16];
#pragma unroll
    for (int i = 0; i < 8; ++i) {
        const int mi0 = mBase + i + (hgrp << 3);
        if (mi0 < M) {
            const int bb = mi0 / OHOW;
            const int rr = mi0 - bb * OHOW;
            const size_t off = ((size_t)bb * OC + n0) * OHOW + rr;
            y[off]                     = acc00[i] + bia0;
            y[off + (size_t)16 * OHOW] = acc01[i] + bia1;
        }
        const int mi1 = mi0 + 16;
        if (mi1 < M) {
            const int bb = mi1 / OHOW;
            const int rr = mi1 - bb * OHOW;
            const size_t off = ((size_t)bb * OC + n0) * OHOW + rr;
            y[off]                     = acc10[i] + bia0;
            y[off + (size_t)16 * OHOW] = acc11[i] + bia1;
        }
    }
}

// ---------------------------------------------------------------------------
// Per-channel batch statistics: one block per channel (input NCHW f32).
// stats[c] = mean, stats[C+c] = rsqrt(var+eps), biased variance (training BN).
// ---------------------------------------------------------------------------
__global__ __launch_bounds__(256) void stats_kernel(
    const float* __restrict__ a, float* __restrict__ stats, int B, int C, int HW)
{
    const int c = blockIdx.x;
    const int n = B * HW;
    float s = 0.0f, s2 = 0.0f;
    for (int t = threadIdx.x; t < n; t += blockDim.x) {
        const int b = t / HW;
        const int i = t - b * HW;
        const float v = a[((size_t)b * C + c) * HW + i];
        s += v;
        s2 += v * v;
    }
    __shared__ float sh0[256];
    __shared__ float sh1[256];
    sh0[threadIdx.x] = s;
    sh1[threadIdx.x] = s2;
    __syncthreads();
    for (int st = 128; st > 0; st >>= 1) {
        if ((int)threadIdx.x < st) {
            sh0[threadIdx.x] += sh0[threadIdx.x + st];
            sh1[threadIdx.x] += sh1[threadIdx.x + st];
        }
        __syncthreads();
    }
    if (threadIdx.x == 0) {
        const float mean = sh0[0] / (float)n;
        const float var  = sh1[0] / (float)n - mean * mean;
        stats[c]     = mean;
        stats[C + c] = rsqrtf(var + 1e-5f);
    }
}

// ---------------------------------------------------------------------------
// BN (batch stats) + ReLU: NCHW f32 in -> NHWC f16 out (coalesced stores),
// feeding the next layer's vectorized WMMA gather.
// ---------------------------------------------------------------------------
__global__ __launch_bounds__(256) void bn_relu_f16_nhwc_kernel(
    const float* __restrict__ a, const float* __restrict__ stats,
    const float* __restrict__ g, const float* __restrict__ beta,
    _Float16* __restrict__ out, int C, int HW, int total)
{
    const int idx = blockIdx.x * blockDim.x + threadIdx.x;   // NHWC linear index
    if (idx >= total) return;
    const int b  = idx / (C * HW);
    const int rm = idx - b * C * HW;
    const int hw = rm / C;
    const int c  = rm - hw * C;
    const float v = (a[((size_t)b * C + c) * HW + hw] - stats[c]) * stats[C + c] * g[c] + beta[c];
    out[idx] = (_Float16)(v > 0.0f ? v : 0.0f);
}

// ---------------------------------------------------------------------------
// Final tiny FC: out[b][j] = fc3_b[j] + sum_k h[b][k]*fc3_w[j][k]  (K=32, N=9)
// ---------------------------------------------------------------------------
__global__ __launch_bounds__(256) void fc3_kernel(
    const _Float16* __restrict__ h, const float* __restrict__ w,
    const float* __restrict__ bias, float* __restrict__ out, int B)
{
    const int idx = blockIdx.x * blockDim.x + threadIdx.x;
    if (idx >= B * 9) return;
    const int b = idx / 9;
    const int j = idx - b * 9;
    float s = bias[j];
#pragma unroll
    for (int k = 0; k < 32; ++k) s += (float)h[b * 32 + k] * w[j * 32 + k];
    out[idx] = s;
}

extern "C" void kernel_launch(void* const* d_in, const int* in_sizes, int n_in,
                              void* d_out, int out_size, void* d_ws, size_t ws_size,
                              hipStream_t stream) {
    const float* x       = (const float*)d_in[0];
    const float* conv1_w = (const float*)d_in[1];
    const float* conv1_b = (const float*)d_in[2];
    const float* bn1_g   = (const float*)d_in[3];
    const float* bn1_b   = (const float*)d_in[4];
    const float* conv2_w = (const float*)d_in[5];
    const float* conv2_b = (const float*)d_in[6];
    const float* bn2_g   = (const float*)d_in[7];
    const float* bn2_b   = (const float*)d_in[8];
    const float* conv3_w = (const float*)d_in[9];
    const float* conv3_b = (const float*)d_in[10];
    const float* bn3_g   = (const float*)d_in[11];
    const float* bn3_b   = (const float*)d_in[12];
    const float* fc1_w   = (const float*)d_in[13];
    const float* fc1_b   = (const float*)d_in[14];
    const float* bnf1_g  = (const float*)d_in[15];
    const float* bnf1_b  = (const float*)d_in[16];
    const float* fc2_w   = (const float*)d_in[17];
    const float* fc2_b   = (const float*)d_in[18];
    const float* bnf2_g  = (const float*)d_in[19];
    const float* bnf2_b  = (const float*)d_in[20];
    const float* fc3_w   = (const float*)d_in[21];
    const float* fc3_b   = (const float*)d_in[22];
    float* out = (float*)d_out;

    const int B = in_sizes[0] / (3 * 128 * 128);   // 256

    // --- workspace carve-up (256B aligned chunks) ---
    char* cur = (char*)d_ws;
    auto carve = [&](size_t bytes) -> char* {
        char* p = cur;
        cur += (bytes + 255) & ~(size_t)255;
        return p;
    };
    float*     pre32 = (float*)    carve((size_t)B * 32 * 961 * sizeof(float)); // reused by all layers
    _Float16*  act1h = (_Float16*) carve((size_t)B * 961 * 32 * sizeof(_Float16)); // NHWC
    _Float16*  act2h = (_Float16*) carve((size_t)B * 100 * 64 * sizeof(_Float16)); // NHWC
    _Float16*  act3h = (_Float16*) carve((size_t)B * 64 * 64  * sizeof(_Float16)); // NHWC
    _Float16*  fca1h = (_Float16*) carve((size_t)B * 256      * sizeof(_Float16));
    _Float16*  fca2h = (_Float16*) carve((size_t)B * 32       * sizeof(_Float16));
    _Float16*  wp1   = (_Float16*) carve((size_t)32 * 192   * sizeof(_Float16));
    _Float16*  wp2   = (_Float16*) carve((size_t)64 * 512   * sizeof(_Float16));
    _Float16*  wp3   = (_Float16*) carve((size_t)64 * 576   * sizeof(_Float16));
    _Float16*  wp4   = (_Float16*) carve((size_t)256 * 4096 * sizeof(_Float16));
    _Float16*  wp5   = (_Float16*) carve((size_t)32 * 256   * sizeof(_Float16));
    float*     stats = (float*)    carve(512 * sizeof(float));

    const dim3 blk(256);
    auto nb = [](int total) { return dim3((unsigned)((total + 255) / 256)); };

    // --- pack binarized weights into WMMA B-fragment layout ---
    // C2 remap makes the GEMM k-order channel-fastest to match NHWC gathers.
    pack_w_kernel<<<nb(32 * 192),   blk, 0, stream>>>(conv1_w, wp1, 32, 192, 1);    // conv1: NCHW k-order
    pack_w_kernel<<<nb(64 * 512),   blk, 0, stream>>>(conv2_w, wp2, 64, 512, 32);   // k = khkw*32 + c
    pack_w_kernel<<<nb(64 * 576),   blk, 0, stream>>>(conv3_w, wp3, 64, 576, 64);   // k = khkw*64 + c
    pack_w_kernel<<<nb(256 * 4096), blk, 0, stream>>>(fc1_w,   wp4, 256, 4096, 64); // k = hw*64 + oc (NHWC flatten)
    pack_w_kernel<<<nb(32 * 256),   blk, 0, stream>>>(fc2_w,   wp5, 32, 256, 1);    // identity
    // ---- Layer 1: conv 3->32, 8x8 s4, 128 -> 31 ----
    {
        const int M = B * 31 * 31;
        dim3 grid((M + 255) / 256, 1);
        binconv_wmma<true, 3, 128, 128, 8, 8, 4, 31, 31, 32>
            <<<grid, blk, 0, stream>>>((const void*)x, wp1, conv1_b, pre32, B);
        stats_kernel<<<dim3(32), blk, 0, stream>>>(pre32, stats, B, 32, 961);
        const int total = B * 32 * 961;
        bn_relu_f16_nhwc_kernel<<<nb(total), blk, 0, stream>>>(pre32, stats, bn1_g, bn1_b, act1h, 32, 961, total);
    }
    // ---- Layer 2: conv 32->64, 4x4 s3, 31 -> 10 ----
    {
        const int M = B * 100;
        dim3 grid((M + 255) / 256, 2);
        binconv_wmma<false, 32, 31, 31, 4, 4, 3, 10, 10, 64>
            <<<grid, blk, 0, stream>>>((const void*)act1h, wp2, conv2_b, pre32, B);
        stats_kernel<<<dim3(64), blk, 0, stream>>>(pre32, stats, B, 64, 100);
        const int total = B * 64 * 100;
        bn_relu_f16_nhwc_kernel<<<nb(total), blk, 0, stream>>>(pre32, stats, bn2_g, bn2_b, act2h, 64, 100, total);
    }
    // ---- Layer 3: conv 64->64, 3x3 s1, 10 -> 8 ----
    {
        const int M = B * 64;
        dim3 grid((M + 255) / 256, 2);
        binconv_wmma<false, 64, 10, 10, 3, 3, 1, 8, 8, 64>
            <<<grid, blk, 0, stream>>>((const void*)act2h, wp3, conv3_b, pre32, B);
        stats_kernel<<<dim3(64), blk, 0, stream>>>(pre32, stats, B, 64, 64);
        const int total = B * 64 * 64;
        bn_relu_f16_nhwc_kernel<<<nb(total), blk, 0, stream>>>(pre32, stats, bn3_g, bn3_b, act3h, 64, 64, total);
    }
    // ---- FC1: 4096 -> 256 (consumes NHWC-flattened conv3 output) ----
    {
        dim3 grid((B + 255) / 256, 8);
        binconv_wmma<false, 4096, 1, 1, 1, 1, 1, 1, 1, 256>
            <<<grid, blk, 0, stream>>>((const void*)act3h, wp4, fc1_b, pre32, B);
        stats_kernel<<<dim3(256), blk, 0, stream>>>(pre32, stats, B, 256, 1);
        const int total = B * 256;
        bn_relu_f16_nhwc_kernel<<<nb(total), blk, 0, stream>>>(pre32, stats, bnf1_g, bnf1_b, fca1h, 256, 1, total);
    }
    // ---- FC2: 256 -> 32 ----
    {
        dim3 grid((B + 255) / 256, 1);
        binconv_wmma<false, 256, 1, 1, 1, 1, 1, 1, 1, 32>
            <<<grid, blk, 0, stream>>>((const void*)fca1h, wp5, fc2_b, pre32, B);
        stats_kernel<<<dim3(32), blk, 0, stream>>>(pre32, stats, B, 32, 1);
        const int total = B * 32;
        bn_relu_f16_nhwc_kernel<<<nb(total), blk, 0, stream>>>(pre32, stats, bnf2_g, bnf2_b, fca2h, 32, 1, total);
    }
    // ---- FC3: 32 -> 9 (real weights, scalar) ----
    fc3_kernel<<<nb(B * 9), blk, 0, stream>>>(fca2h, fc3_w, fc3_b, out, B);
}
